// Transformer_50173807952516
// MI455X (gfx1250) — matrix operand order repaired and checked by reading, not compiled, not run on hardware
//
#include <hip/hip_runtime.h>

// ---------------------------------------------------------------------------
// Transformer encoder-decoder forward for MI455X (gfx1250, wave32, WMMA).
// - All matrix math: v_wmma_f32_16x16x32_bf16 (bf16 in, f32 accumulate).
// - Weights converted to bf16 once per launch; activations carry bf16 shadows.
// - LDS tile staging uses GLOBAL_LOAD_ASYNC_TO_LDS_B128 (ASYNCcnt) when the
//   builtin exists, with a synchronous uint4 fallback.
// ---------------------------------------------------------------------------

#if defined(__has_builtin)
#  if __has_builtin(__builtin_amdgcn_global_load_async_to_lds_b128)
#    define HAVE_ASYNC_LDS 1
#  endif
#endif
#ifndef HAVE_ASYNC_LDS
#  define HAVE_ASYNC_LDS 0
#endif

typedef __attribute__((ext_vector_type(16))) __bf16 v16bf;
typedef __attribute__((ext_vector_type(8)))  float  v8f;

union FragA {
  v16bf v;
  unsigned int   d[8];
  unsigned short u[16];
};

static __device__ __forceinline__ v8f wmma_bf16(const FragA& a, const FragA& b, v8f c) {
  return __builtin_amdgcn_wmma_f32_16x16x32_bf16(
      /*neg_a=*/false, a.v, /*neg_b=*/false, b.v,
      /*c_mod=*/(short)0, c, /*reuse_a=*/false, /*reuse_b=*/false);
}

static __device__ __forceinline__ unsigned short f2bf(float f) {
  unsigned int u = __float_as_uint(f);
  u += 0x7FFFu + ((u >> 16) & 1u);        // round-to-nearest-even
  return (unsigned short)(u >> 16);
}

static __device__ __forceinline__ v8f v8f_zero() {
  v8f z = {0.f, 0.f, 0.f, 0.f, 0.f, 0.f, 0.f, 0.f};
  return z;
}

#if HAVE_ASYNC_LDS
// Builtin signature (leaked by the round-2 diagnostic): param 1 is a pointer
// to a GCC-vector int4 in the device (AS1) address space; LDS side is AS3.
typedef int v4i_g __attribute__((vector_size(16)));
static __device__ __forceinline__ void async_cp16(const void* g, void* l) {
  __builtin_amdgcn_global_load_async_to_lds_b128(
      (__attribute__((address_space(1))) v4i_g*)g,
      (__attribute__((address_space(3))) v4i_g*)l, 0, 0);
}
static __device__ __forceinline__ void wait_async0() {
#  if __has_builtin(__builtin_amdgcn_s_wait_asynccnt)
  __builtin_amdgcn_s_wait_asynccnt(0);
#  else
  asm volatile("s_wait_asynccnt 0" ::: "memory");
#  endif
}
#endif

// ---------------------------------------------------------------------------
// f32 -> bf16 conversion (weights, once per launch)
// ---------------------------------------------------------------------------
__global__ __launch_bounds__(256) void cvt_kernel(
    const float* __restrict__ in, unsigned short* __restrict__ out, int n) {
  int i = blockIdx.x * 256 + threadIdx.x;
  if (i < n) out[i] = f2bf(in[i]);
}

// ---------------------------------------------------------------------------
// Embedding + positional encoding: writes f32 master + bf16 shadow. D = 512.
// ---------------------------------------------------------------------------
__global__ __launch_bounds__(256) void embed_kernel(
    const int* __restrict__ tok, const float* __restrict__ emb,
    const float* __restrict__ pe, float* __restrict__ outf,
    unsigned short* __restrict__ outb, int L, int total) {
  int i = blockIdx.x * 256 + threadIdx.x;
  if (i >= total) return;
  int d   = i & 511;
  int row = i >> 9;
  int pos = row % L;
  float v = emb[(size_t)tok[row] * 512 + d] + pe[(size_t)pos * 512 + d];
  outf[i] = v;
  outb[i] = f2bf(v);
}

// ---------------------------------------------------------------------------
// Fused residual + LayerNorm over D=512; writes f32 master + bf16 shadow.
// ---------------------------------------------------------------------------
__global__ __launch_bounds__(128) void ln_kernel(
    const float* __restrict__ x, const float* __restrict__ res,
    const float* __restrict__ sc, const float* __restrict__ bs,
    float* __restrict__ outf, unsigned short* __restrict__ outb) {
  const int row = blockIdx.x;
  const int t   = threadIdx.x;
  __shared__ float sdat[4];
  float v[4];
  float sum = 0.f;
#pragma unroll
  for (int i = 0; i < 4; ++i) {
    int d = i * 128 + t;
    v[i] = x[(size_t)row * 512 + d] + res[(size_t)row * 512 + d];
    sum += v[i];
  }
#pragma unroll
  for (int off = 16; off >= 1; off >>= 1) sum += __shfl_xor(sum, off, 32);
  if ((t & 31) == 0) sdat[t >> 5] = sum;
  __syncthreads();
  float mean = (sdat[0] + sdat[1] + sdat[2] + sdat[3]) * (1.f / 512.f);
  __syncthreads();
  float var = 0.f;
#pragma unroll
  for (int i = 0; i < 4; ++i) { float dd = v[i] - mean; var += dd * dd; }
#pragma unroll
  for (int off = 16; off >= 1; off >>= 1) var += __shfl_xor(var, off, 32);
  if ((t & 31) == 0) sdat[t >> 5] = var;
  __syncthreads();
  float rstd = rsqrtf((sdat[0] + sdat[1] + sdat[2] + sdat[3]) * (1.f / 512.f) + 1e-5f);
#pragma unroll
  for (int i = 0; i < 4; ++i) {
    int d = i * 128 + t;
    float o = (v[i] - mean) * rstd * sc[d] + bs[d];
    outf[(size_t)row * 512 + d] = o;
    outb[(size_t)row * 512 + d] = f2bf(o);
  }
}

// ---------------------------------------------------------------------------
// WMMA GEMM, all-bf16 operands: out[M,N] = A[M,K] @ W[K,N] + bias[N]
// Block tile 64(M) x 128(N), BK = 32; 8 waves in 2x4 grid, 32x32 per wave.
// A staged row-major [64][40] (80B rows); W staged k-major [32][136] (272B
// rows). Both staged with async global->LDS b128 copies when available.
// ---------------------------------------------------------------------------
template <int OBF16, int RELU>
__global__ __launch_bounds__(256) void gemm_bf(
    const unsigned short* __restrict__ A, const unsigned short* __restrict__ W,
    const float* __restrict__ bias, void* __restrict__ outv,
    int M, int N, int K) {
  __shared__ __align__(16) unsigned short As[64][40];
  __shared__ __align__(16) unsigned short Bs[32][136];

  const int tid  = threadIdx.x;
  const int lane = tid & 31;
  const int wv   = tid >> 5;
  const int lr   = lane & 15;
  const int hl   = lane >> 4;        // half-lane group (0/1)
  const int wm   = wv >> 2;          // 0..1
  const int wn   = wv & 3;           // 0..3
  const int m0   = blockIdx.y * 64;
  const int n0   = blockIdx.x * 128;

  v8f acc[2][2];
  acc[0][0] = v8f_zero(); acc[0][1] = v8f_zero();
  acc[1][0] = v8f_zero(); acc[1][1] = v8f_zero();

  for (int k0 = 0; k0 < K; k0 += 32) {
    if (k0 + 32 < K)   // prefetch next W tile (global_prefetch_b8)
      __builtin_prefetch(&W[(size_t)(k0 + 32 + (tid & 31)) * N + n0 + ((tid >> 5) << 4)], 0, 1);

    // Stage A tile (64 rows x 32 k): 256 chunks of 8 bf16, 1 per thread
    {
      int row = tid >> 2, cg = (tid & 3) << 3;
      const unsigned short* ga = &A[(size_t)(m0 + row) * K + k0 + cg];
#if HAVE_ASYNC_LDS
      async_cp16(ga, &As[row][cg]);
#else
      *(uint4*)&As[row][cg] = *(const uint4*)ga;
#endif
    }
    // Stage W tile k-major (32 k x 128 n): 512 chunks, 2 per thread
#pragma unroll
    for (int i = 0; i < 2; ++i) {
      int c  = tid + 256 * i;
      int kr = c >> 4, ng = (c & 15) << 3;
      const unsigned short* gw = &W[(size_t)(k0 + kr) * N + n0 + ng];
#if HAVE_ASYNC_LDS
      async_cp16(gw, &Bs[kr][ng]);
#else
      *(uint4*)&Bs[kr][ng] = *(const uint4*)gw;
#endif
    }
#if HAVE_ASYNC_LDS
    wait_async0();
#endif
    __syncthreads();

    FragA a0, a1, b0, b1;
#pragma unroll
    for (int p = 0; p < 8; ++p) {
      // A-layout K mapping: k = (p/4)*16 + (p%4)*2 + half*8 (pairs contiguous)
      int ka = ((p >> 2) << 4) + ((p & 3) << 1) + (hl << 3);
      a0.d[p] = *(const unsigned int*)&As[wm * 32 + lr][ka];
      a1.d[p] = *(const unsigned int*)&As[wm * 32 + 16 + lr][ka];
    }
#pragma unroll
    for (int e = 0; e < 16; ++e) {
      // B-layout K mapping: k = (e/8)*16 + half*8 + e%8; column = n-tile base + lr
      int k = ((e >> 3) << 4) + (hl << 3) + (e & 7);
      b0.u[e] = Bs[k][wn * 32 + lr];
      b1.u[e] = Bs[k][wn * 32 + 16 + lr];
    }
    acc[0][0] = wmma_bf16(a0, b0, acc[0][0]);
    acc[0][1] = wmma_bf16(a0, b1, acc[0][1]);
    acc[1][0] = wmma_bf16(a1, b0, acc[1][0]);
    acc[1][1] = wmma_bf16(a1, b1, acc[1][1]);
    __syncthreads();
  }

  // Epilogue: C layout -> lane lr = column, row = r + 8*hl within 16-tile
#pragma unroll
  for (int mt = 0; mt < 2; ++mt)
#pragma unroll
    for (int nt = 0; nt < 2; ++nt) {
      int n = n0 + wn * 32 + nt * 16 + lr;
      float bvv = bias[n];
#pragma unroll
      for (int r = 0; r < 8; ++r) {
        float val = acc[mt][nt][r] + bvv;
        if (RELU) val = fmaxf(val, 0.f);
        size_t idx = (size_t)(m0 + wm * 32 + mt * 16 + r + hl * 8) * N + n;
        if (OBF16) ((unsigned short*)outv)[idx] = f2bf(val);
        else       ((float*)outv)[idx] = val;
      }
    }
}

// ---------------------------------------------------------------------------
// Flash attention, one wave per (b, h, 16-query tile). dk = 64, D = 512, H = 8.
// Q/K/V bf16 [B,L,D]; 32 keys per iteration: S via 4 WMMA, ctx via 4 WMMA.
// Online softmax via wave32 16-lane shuffles. Output: bf16 ctx shadow.
// ---------------------------------------------------------------------------
#define AT_D  512
#define AT_H  8
#define AT_DK 64

__global__ __launch_bounds__(128) void attn_kernel(
    const unsigned short* __restrict__ Q, const unsigned short* __restrict__ Kt,
    const unsigned short* __restrict__ Vt, const int* __restrict__ mask,
    unsigned short* __restrict__ out, int Lq, int Lk,
    int mask_b_stride, int mask_q_stride) {
  __shared__ __align__(16) unsigned short sP[4][16][34];  // P tile per wave
  __shared__ __align__(16) unsigned short sV[4][32][72];  // V tile per wave

  const int tid  = threadIdx.x;
  const int wv   = tid >> 5;
  const int lane = tid & 31;
  const int lr   = lane & 15;
  const int hl   = lane >> 4;
  const int nQT  = Lq >> 4;
  const int wid  = blockIdx.x * 4 + wv;
  const int b    = wid / (AT_H * nQT);
  const int h    = (wid / nQT) % AT_H;
  const int qt   = wid % nQT;

  const unsigned short* qrow = Q + (size_t)(b * Lq + qt * 16 + lr) * AT_D + h * AT_DK;
  FragA aq[2];
#pragma unroll
  for (int s = 0; s < 2; ++s)
#pragma unroll
    for (int p = 0; p < 8; ++p) {
      int k = s * 32 + ((p >> 2) << 4) + ((p & 3) << 1) + (hl << 3);
      aq[s].d[p] = *(const unsigned int*)&qrow[k];
    }

  v8f accv[4];
  accv[0] = v8f_zero(); accv[1] = v8f_zero(); accv[2] = v8f_zero(); accv[3] = v8f_zero();
  float mrun[8], lrun[8];
#pragma unroll
  for (int r = 0; r < 8; ++r) { mrun[r] = -1e30f; lrun[r] = 0.f; }

  const int* mbase = mask + (size_t)b * mask_b_stride;

  for (int kb = 0; kb < Lk; kb += 32) {
    // ---- S = Q @ K^T for two 16-key subtiles -----------------------------
    v8f s0 = v8f_zero(), s1 = v8f_zero();
#pragma unroll
    for (int t = 0; t < 2; ++t) {
      const unsigned short* krow =
          Kt + (size_t)(b * Lk + kb + t * 16 + lr) * AT_D + h * AT_DK;
      FragA bk0, bk1;
#pragma unroll
      for (int p = 0; p < 8; ++p) {
        int kk = ((p >> 2) << 4) + (hl << 3) + ((p & 3) << 1);
        bk0.d[p] = *(const unsigned int*)&krow[kk];        // dk 0..31
        bk1.d[p] = *(const unsigned int*)&krow[32 + kk];   // dk 32..63
      }
      if (t == 0) { s0 = wmma_bf16(aq[0], bk0, s0); s0 = wmma_bf16(aq[1], bk1, s0); }
      else        { s1 = wmma_bf16(aq[0], bk0, s1); s1 = wmma_bf16(aq[1], bk1, s1); }
    }

    // ---- Stage V tile (32 keys x 64), one row per lane -------------------
    {
      const unsigned short* vrow =
          Vt + (size_t)(b * Lk + kb + lane) * AT_D + h * AT_DK;
#pragma unroll
      for (int c = 0; c < 64; c += 8) {
#if HAVE_ASYNC_LDS
        async_cp16(&vrow[c], &sV[wv][lane][c]);
#else
        *(uint4*)&sV[wv][lane][c] = *(const uint4*)&vrow[c];
#endif
      }
    }

    // ---- Scale, mask, online softmax update ------------------------------
#pragma unroll
    for (int r = 0; r < 8; ++r) {
      float f0 = s0[r] * 0.125f;   // 1/sqrt(64)
      float f1 = s1[r] * 0.125f;
      int q = qt * 16 + r + hl * 8;
      const int* mrow = mbase + (size_t)q * mask_q_stride;
      if (mrow[kb + lr] == 0)      f0 = -1e9f;
      if (mrow[kb + 16 + lr] == 0) f1 = -1e9f;
      float mx = fmaxf(f0, f1);
      mx = fmaxf(mx, __shfl_xor(mx, 1, 32));
      mx = fmaxf(mx, __shfl_xor(mx, 2, 32));
      mx = fmaxf(mx, __shfl_xor(mx, 4, 32));
      mx = fmaxf(mx, __shfl_xor(mx, 8, 32));
      float mnew = fmaxf(mrun[r], mx);
      float p0 = __expf(f0 - mnew);
      float p1 = __expf(f1 - mnew);
      float rs = p0 + p1;
      rs += __shfl_xor(rs, 1, 32);
      rs += __shfl_xor(rs, 2, 32);
      rs += __shfl_xor(rs, 4, 32);
      rs += __shfl_xor(rs, 8, 32);
      float corr = __expf(mrun[r] - mnew);
      lrun[r] = lrun[r] * corr + rs;
      mrun[r] = mnew;
      sP[wv][r + hl * 8][lr]      = f2bf(p0);
      sP[wv][r + hl * 8][16 + lr] = f2bf(p1);
      accv[0][r] *= corr; accv[1][r] *= corr;
      accv[2][r] *= corr; accv[3][r] *= corr;
    }
#if HAVE_ASYNC_LDS
    wait_async0();
#endif
    __syncthreads();

    // ---- ctx += P @ V ----------------------------------------------------
    FragA ap;
#pragma unroll
    for (int p = 0; p < 8; ++p) {
      int k = ((p >> 2) << 4) + ((p & 3) << 1) + (hl << 3);
      ap.d[p] = *(const unsigned int*)&sP[wv][lr][k];
    }
#pragma unroll
    for (int nt = 0; nt < 4; ++nt) {
      FragA bvf;
#pragma unroll
      for (int p = 0; p < 8; ++p) {
        int key = ((p >> 2) << 4) + (hl << 3) + ((p & 3) << 1);
        bvf.u[2 * p]     = sV[wv][key][nt * 16 + lr];
        bvf.u[2 * p + 1] = sV[wv][key + 1][nt * 16 + lr];
      }
      accv[nt] = wmma_bf16(ap, bvf, accv[nt]);
    }
    __syncthreads();
  }

  // ---- Normalize and write bf16 ctx (concat layout [B, Lq, D]) -----------
#pragma unroll
  for (int nt = 0; nt < 4; ++nt)
#pragma unroll
    for (int r = 0; r < 8; ++r) {
      size_t o = (size_t)(b * Lq + qt * 16 + r + hl * 8) * AT_D + h * AT_DK + nt * 16 + lr;
      out[o] = f2bf(accv[nt][r] / lrun[r]);
    }
}

// ---------------------------------------------------------------------------
// Host-side sequencing helpers
// ---------------------------------------------------------------------------
static void run_gemm(const unsigned short* A, const unsigned short* W,
                     const float* bias, void* out, int M, int N, int K,
                     int obf, int relu, hipStream_t s) {
  dim3 grid((unsigned)(N / 128), (unsigned)(M / 64));
  if (obf && relu)       gemm_bf<1, 1><<<grid, 256, 0, s>>>(A, W, bias, out, M, N, K);
  else if (obf)          gemm_bf<1, 0><<<grid, 256, 0, s>>>(A, W, bias, out, M, N, K);
  else if (relu)         gemm_bf<0, 1><<<grid, 256, 0, s>>>(A, W, bias, out, M, N, K);
  else                   gemm_bf<0, 0><<<grid, 256, 0, s>>>(A, W, bias, out, M, N, K);
}

static void run_attn(const unsigned short* Q, const unsigned short* K,
                     const unsigned short* V, const int* mask, unsigned short* out,
                     int Lq, int Lk, int mbs, int mqs, hipStream_t s) {
  int blocks = (4 * 8 * (Lq / 16)) / 4;   // B*H*qtiles / 4 waves per block
  attn_kernel<<<blocks, 128, 0, s>>>(Q, K, V, mask, out, Lq, Lk, mbs, mqs);
}

static void run_ln(const float* x, const float* r, const float* sc, const float* bs,
                   float* outf, unsigned short* outb, int rows, hipStream_t s) {
  ln_kernel<<<rows, 128, 0, s>>>(x, r, sc, bs, outf, outb);
}

static void run_cvt(const float* in, unsigned short* out, int n, hipStream_t s) {
  cvt_kernel<<<(n + 255) / 256, 256, 0, s>>>(in, out, n);
}

extern "C" void kernel_launch(void* const* d_in, const int* in_sizes, int n_in,
                              void* d_out, int out_size, void* d_ws, size_t ws_size,
                              hipStream_t stream) {
  (void)in_sizes; (void)n_in; (void)out_size; (void)ws_size;
  const int B = 4, LS = 512, LT = 512, D = 512, FF = 2048, NL = 2, VT = 32000;
  const int M = B * LS;   // 2048 rows (encoder and decoder)

  const int*   src        = (const int*)d_in[0];
  const int*   tgt        = (const int*)d_in[1];
  const int*   src_mask   = (const int*)d_in[2];
  const int*   tgt_mask   = (const int*)d_in[3];
  const float* src_emb    = (const float*)d_in[4];
  const float* tgt_emb    = (const float*)d_in[5];
  const float* pe         = (const float*)d_in[6];
  const float* enc_attn_w = (const float*)d_in[7];
  const float* enc_attn_b = (const float*)d_in[8];
  const float* enc_ff_w1  = (const float*)d_in[9];
  const float* enc_ff_b1  = (const float*)d_in[10];
  const float* enc_ff_w2  = (const float*)d_in[11];
  const float* enc_ff_b2  = (const float*)d_in[12];
  const float* enc_ln_s   = (const float*)d_in[13];
  const float* enc_ln_b   = (const float*)d_in[14];
  const float* dec_attn_w = (const float*)d_in[15];
  const float* dec_attn_b = (const float*)d_in[16];
  const float* dec_ff_w1  = (const float*)d_in[17];
  const float* dec_ff_b1  = (const float*)d_in[18];
  const float* dec_ff_w2  = (const float*)d_in[19];
  const float* dec_ff_b2  = (const float*)d_in[20];
  const float* dec_ln_s   = (const float*)d_in[21];
  const float* dec_ln_b   = (const float*)d_in[22];
  const float* out_w      = (const float*)d_in[23];
  const float* out_b      = (const float*)d_in[24];

  // Workspace layout (~88 MB)
  char* ws = (char*)d_ws;
  float*          xa   = (float*)(ws);                          //  4 MB f32 act
  float*          xb   = (float*)(ws + (4ull  << 20));          //  4 MB f32 sub out
  unsigned short* xab  = (unsigned short*)(ws + (8ull  << 20)); //  2 MB bf16 act
  unsigned short* eob  = (unsigned short*)(ws + (10ull << 20)); //  2 MB enc out bf16
  unsigned short* qb   = (unsigned short*)(ws + (12ull << 20)); //  2 MB Q
  unsigned short* kb   = (unsigned short*)(ws + (14ull << 20)); //  2 MB K
  unsigned short* vb   = (unsigned short*)(ws + (16ull << 20)); //  2 MB V
  unsigned short* ctxb = (unsigned short*)(ws + (18ull << 20)); //  2 MB ctx
  unsigned short* ffmb = (unsigned short*)(ws + (20ull << 20)); //  8 MB FF hidden
  unsigned short* wEA  = (unsigned short*)(ws + (28ull << 20)); //  4 MB enc attn w
  unsigned short* wEF1 = (unsigned short*)(ws + (32ull << 20)); //  4 MB enc ff1 w
  unsigned short* wEF2 = (unsigned short*)(ws + (36ull << 20)); //  4 MB enc ff2 w
  unsigned short* wDA  = (unsigned short*)(ws + (40ull << 20)); //  8 MB dec attn w
  unsigned short* wDF1 = (unsigned short*)(ws + (48ull << 20)); //  4 MB dec ff1 w
  unsigned short* wDF2 = (unsigned short*)(ws + (52ull << 20)); //  4 MB dec ff2 w
  unsigned short* wOUT = (unsigned short*)(ws + (56ull << 20)); // 31.25 MB out_w

  const size_t DD = (size_t)D * D;

  // --- one-time (per launch) weight conversion to bf16 ---
  run_cvt(enc_attn_w, wEA,  NL * 4 * D * D, stream);
  run_cvt(enc_ff_w1,  wEF1, NL * D * FF,    stream);
  run_cvt(enc_ff_w2,  wEF2, NL * FF * D,    stream);
  run_cvt(dec_attn_w, wDA,  NL * 8 * D * D, stream);
  run_cvt(dec_ff_w1,  wDF1, NL * D * FF,    stream);
  run_cvt(dec_ff_w2,  wDF2, NL * FF * D,    stream);
  run_cvt(out_w,      wOUT, D * VT,         stream);

  // ======================== Encoder ========================
  embed_kernel<<<(M * D) / 256, 256, 0, stream>>>(src, src_emb, pe, xa, xab, LS, M * D);
  for (int l = 0; l < NL; ++l) {
    const unsigned short* w = wEA + (size_t)l * 4 * DD;
    const float*          bp = enc_attn_b + (size_t)l * 4 * D;
    run_gemm(xab, w + 0 * DD, bp + 0 * D, qb, M, D, D, 1, 0, stream);
    run_gemm(xab, w + 1 * DD, bp + 1 * D, kb, M, D, D, 1, 0, stream);
    run_gemm(xab, w + 2 * DD, bp + 2 * D, vb, M, D, D, 1, 0, stream);
    run_attn(qb, kb, vb, src_mask, ctxb, LS, LS, LS, 0, stream);
    run_gemm(ctxb, w + 3 * DD, bp + 3 * D, xb, M, D, D, 0, 0, stream);
    run_ln(xa, xb, enc_ln_s + (size_t)(l * 2 + 0) * D, enc_ln_b + (size_t)(l * 2 + 0) * D,
           xa, xab, M, stream);
    run_gemm(xab, wEF1 + (size_t)l * D * FF, enc_ff_b1 + (size_t)l * FF, ffmb, M, FF, D, 1, 1, stream);
    run_gemm(ffmb, wEF2 + (size_t)l * FF * D, enc_ff_b2 + (size_t)l * D, xb, M, D, FF, 0, 0, stream);
    unsigned short* lnb16 = (l == NL - 1) ? eob : xab;
    run_ln(xa, xb, enc_ln_s + (size_t)(l * 2 + 1) * D, enc_ln_b + (size_t)(l * 2 + 1) * D,
           xa, lnb16, M, stream);
  }

  // ======================== Decoder ========================
  embed_kernel<<<(M * D) / 256, 256, 0, stream>>>(tgt, tgt_emb, pe, xa, xab, LT, M * D);
  for (int l = 0; l < NL; ++l) {
    const unsigned short* w = wDA + (size_t)l * 8 * DD;
    const float*          bp = dec_attn_b + (size_t)l * 8 * D;
    // --- masked self attention ---
    run_gemm(xab, w + 0 * DD, bp + 0 * D, qb, M, D, D, 1, 0, stream);
    run_gemm(xab, w + 1 * DD, bp + 1 * D, kb, M, D, D, 1, 0, stream);
    run_gemm(xab, w + 2 * DD, bp + 2 * D, vb, M, D, D, 1, 0, stream);
    run_attn(qb, kb, vb, tgt_mask, ctxb, LT, LT, LT * LT, LT, stream);
    run_gemm(ctxb, w + 3 * DD, bp + 3 * D, xb, M, D, D, 0, 0, stream);
    run_ln(xa, xb, dec_ln_s + (size_t)(l * 3 + 0) * D, dec_ln_b + (size_t)(l * 3 + 0) * D,
           xa, xab, M, stream);
    // --- cross attention (K/V from encoder output) ---
    run_gemm(xab, w + 4 * DD, bp + 4 * D, qb, M, D, D, 1, 0, stream);
    run_gemm(eob, w + 5 * DD, bp + 5 * D, kb, M, D, D, 1, 0, stream);
    run_gemm(eob, w + 6 * DD, bp + 6 * D, vb, M, D, D, 1, 0, stream);
    run_attn(qb, kb, vb, src_mask, ctxb, LT, LS, LS, 0, stream);
    run_gemm(ctxb, w + 7 * DD, bp + 7 * D, xb, M, D, D, 0, 0, stream);
    run_ln(xa, xb, dec_ln_s + (size_t)(l * 3 + 1) * D, dec_ln_b + (size_t)(l * 3 + 1) * D,
           xa, xab, M, stream);
    // --- feed forward ---
    run_gemm(xab, wDF1 + (size_t)l * D * FF, dec_ff_b1 + (size_t)l * FF, ffmb, M, FF, D, 1, 1, stream);
    run_gemm(ffmb, wDF2 + (size_t)l * FF * D, dec_ff_b2 + (size_t)l * D, xb, M, D, FF, 0, 0, stream);
    run_ln(xa, xb, dec_ln_s + (size_t)(l * 3 + 2) * D, dec_ln_b + (size_t)(l * 3 + 2) * D,
           xa, xab, M, stream);
  }

  // ======================== Logits ========================
  run_gemm(xab, wOUT, out_b, (float*)d_out, M, VT, D, 0, 0, stream);
}
